// W4A8Layer_72559177499042
// MI455X (gfx1250) — 8-bit matrix operand order repaired and checked by reading, compiled, not run on hardware
//
#include <hip/hip_runtime.h>

typedef __attribute__((ext_vector_type(4))) int   v4i;
typedef __attribute__((ext_vector_type(8))) int   v8i;
typedef __attribute__((ext_vector_type(8))) float v8f;

#define MDIM 2048
#define NDIM 4096
#define KDIM 4096
#define KT64 (KDIM / 64)
#define NGROUPS (KDIM / 128)
#define FRAGB 1024
#define ROWB (KT64 * FRAGB)

__global__ __launch_bounds__(256) void w4a8_pack_a(const int* __restrict__ a,
                                                   unsigned* __restrict__ ap) {
  unsigned tid  = blockIdx.x * 256u + threadIdx.x;
  unsigned v    = tid & 7u;
  unsigned lane = (tid >> 3) & 31u;
  unsigned frag = tid >> 8;
  unsigned kt   = frag & (KT64 - 1);
  unsigned mt   = frag >> 6;
  unsigned m    = mt * 16u + (lane & 15u);
  unsigned half = lane >> 4;
  unsigned k    = kt * 64u + (v >> 1) * 16u + (v & 1u) * 4u + half * 8u;
  v4i x = *(const v4i*)(a + (size_t)m * KDIM + k);
  unsigned d = (x.x & 0xFFu) | ((x.y & 0xFFu) << 8) |
               ((x.z & 0xFFu) << 16) | ((x.w & 0xFFu) << 24);
  ap[tid] = d;
}

__global__ __launch_bounds__(256) void w4a8_pack_b(const int* __restrict__ bq,
                                                   unsigned* __restrict__ bp) {
  unsigned tid  = blockIdx.x * 256u + threadIdx.x;
  unsigned lane = tid & 31u;
  unsigned v    = (tid >> 5) & 7u;
  unsigned frag = tid >> 8;
  unsigned kt   = frag & (KT64 - 1);
  unsigned nt   = frag >> 6;
  unsigned n    = nt * 16u + (lane & 15u);
  unsigned half = lane >> 4;
  unsigned k    = kt * 64u + (v >> 1) * 16u + (v & 1u) * 4u + half * 8u;
  unsigned d = 0;
#pragma unroll
  for (int i = 0; i < 4; ++i) {
    int q = bq[(size_t)(k + i) * NDIM + n];
    d |= ((unsigned)((q - 8) & 0xFF)) << (8 * i);
  }
  bp[(size_t)frag * 256u + lane * 8u + v] = d;
}

__global__ __launch_bounds__(256, 1) void w4a8_wmma_gemm(
    const char* __restrict__ paB, const char* __restrict__ pbB,
    const float* __restrict__ gscale, const float* __restrict__ tscale,
    float* __restrict__ out) {
  const int lane = threadIdx.x & 31;
  const int wave = threadIdx.x >> 5;
  const int wx = wave & 3;
  const int wy = wave >> 2;
  const int mt0 = blockIdx.y * 8 + wy * 4;
  const int nt0 = blockIdx.x * 8 + wx * 2;
  const int col = lane & 15;

  const unsigned a0 = (unsigned)mt0 * ROWB + (unsigned)lane * 32u;
  const unsigned b0 = (unsigned)nt0 * ROWB + (unsigned)lane * 32u;

  v8f facc[4][2] = {};

#pragma unroll 2
  for (int g = 0; g < NGROUPS; ++g) {
    v8i acc[4][2] = {};
#pragma unroll
    for (int s = 0; s < 2; ++s) {
      const unsigned kob = (unsigned)(g * 2 + s) * (unsigned)FRAGB;
      v8i af[4], bf[2];
#pragma unroll
      for (int i = 0; i < 4; ++i)
        af[i] = *(const v8i*)(paB + (size_t)(a0 + (unsigned)i * ROWB + kob));
#pragma unroll
      for (int j = 0; j < 2; ++j)
        bf[j] = *(const v8i*)(pbB + (size_t)(b0 + (unsigned)j * ROWB + kob));
#pragma unroll
      for (int i = 0; i < 4; ++i)
#pragma unroll
        for (int j = 0; j < 2; ++j)
          acc[i][j] = __builtin_amdgcn_wmma_i32_16x16x64_iu8(
               false, af[i],  true, bf[j], acc[i][j],
               false,  false);
    }
    float s0 = gscale[g * NDIM + nt0 * 16 + col];
    float s1 = gscale[g * NDIM + (nt0 + 1) * 16 + col];
#pragma unroll
    for (int i = 0; i < 4; ++i) {
#pragma unroll
      for (int e = 0; e < 8; ++e) {
        facc[i][0][e] += (float)acc[i][0][e] * s0;
        facc[i][1][e] += (float)acc[i][1][e] * s1;
      }
    }
  }

  const int rowoff = (lane >> 4) * 8;
#pragma unroll
  for (int i = 0; i < 4; ++i) {
    const int mbase = (mt0 + i) * 16 + rowoff;
    float ts[8];
#pragma unroll
    for (int e = 0; e < 8; ++e) ts[e] = tscale[mbase + e];
#pragma unroll
    for (int j = 0; j < 2; ++j) {
      const int n = (nt0 + j) * 16 + col;
#pragma unroll
      for (int e = 0; e < 8; ++e)
        out[(size_t)(mbase + e) * NDIM + n] = facc[i][j][e] * ts[e];
    }
  }
}

extern "C" void kernel_launch(void* const* d_in, const int* in_sizes, int n_in,
                              void* d_out, int out_size, void* d_ws, size_t ws_size,
                              hipStream_t stream) {
  const int*   a  = (const int*)d_in[0];
  const int*   bq = (const int*)d_in[1];
  const float* gs = (const float*)d_in[2];
  const float* ts = (const float*)d_in[3];
  float* out = (float*)d_out;

  unsigned char* ws = (unsigned char*)d_ws;
  unsigned* ap = (unsigned*)ws;
  unsigned* bp = (unsigned*)(ws + (size_t)MDIM * KDIM);

  w4a8_pack_a<<<(MDIM * KDIM / 4) / 256, 256, 0, stream>>>(a, ap);
  w4a8_pack_b<<<(KDIM * NDIM / 4) / 256, 256, 0, stream>>>(bq, bp);

  dim3 grid(NDIM / 128, MDIM / 128);
  w4a8_wmma_gemm<<<grid, 256, 0, stream>>>((const char*)ap, (const char*)bp,
                                           gs, ts, out);
}
